// MirrorDescentLinear_15058155339848
// MI455X (gfx1250) — matrix operand order repaired and checked by reading, not compiled, not run on hardware
//
#include <hip/hip_runtime.h>

// ---------------------------------------------------------------------------
// MirrorDescentLinear on gfx1250 (MI455X):
//   w = (softmax([l0,0,l1]).p2 - .p0) * bf16(scale_group) * mask
//   out = x @ w^T
// GEMM: block tile 128x256 (8 waves, 2x4), wave tile 64x64, K-stage 64,
// double-buffered LDS fed by global_load_async_to_lds_b128 (ASYNCcnt),
// math via v_wmma_f32_16x16x32_bf16.
// ---------------------------------------------------------------------------

typedef __attribute__((ext_vector_type(16))) __bf16 v16bf;
typedef __attribute__((ext_vector_type(8)))  float  v8f;

union BFrag {
    uint4 u[2];   // 32 bytes = 16 bf16
    v16bf v;
};

__device__ __forceinline__ unsigned short f2bf_rne(float f) {
    unsigned int u = __float_as_uint(f);
    unsigned int r = u + 0x7FFFu + ((u >> 16) & 1u);   // round-to-nearest-even
    return (unsigned short)(r >> 16);
}
__device__ __forceinline__ float bf2f(unsigned short h) {
    return __uint_as_float(((unsigned int)h) << 16);
}

// ---- gfx1250 async global->LDS copy helpers (ASYNCcnt-tracked) -------------
// INST_OFFSET is added to BOTH the LDS and the global address, so one base
// pair + offset immediates copies a contiguous run (ISA 08_async_tensor §4.4).
__device__ __forceinline__ void async_copy64(unsigned ldsOff, const char* g) {
    asm volatile(
        "global_load_async_to_lds_b128 %0, %1, off\n\t"
        "global_load_async_to_lds_b128 %0, %1, off offset:16\n\t"
        "global_load_async_to_lds_b128 %0, %1, off offset:32\n\t"
        "global_load_async_to_lds_b128 %0, %1, off offset:48"
        :: "v"(ldsOff), "v"(g) : "memory");
}
__device__ __forceinline__ void async_copy128(unsigned ldsOff, const char* g) {
    asm volatile(
        "global_load_async_to_lds_b128 %0, %1, off\n\t"
        "global_load_async_to_lds_b128 %0, %1, off offset:16\n\t"
        "global_load_async_to_lds_b128 %0, %1, off offset:32\n\t"
        "global_load_async_to_lds_b128 %0, %1, off offset:48\n\t"
        "global_load_async_to_lds_b128 %0, %1, off offset:64\n\t"
        "global_load_async_to_lds_b128 %0, %1, off offset:80\n\t"
        "global_load_async_to_lds_b128 %0, %1, off offset:96\n\t"
        "global_load_async_to_lds_b128 %0, %1, off offset:112"
        :: "v"(ldsOff), "v"(g) : "memory");
}
#define WAIT_ASYNC(n) asm volatile("s_wait_asynccnt " #n ::: "memory")

// ---------------------------------------------------------------------------
// Kernel 1: x (f32) -> x_bf16, 4 elements / thread
// ---------------------------------------------------------------------------
__global__ __launch_bounds__(256) void cvt_x_kernel(const float* __restrict__ x,
                                                    unsigned short* __restrict__ xb,
                                                    int n4) {
    int i = blockIdx.x * blockDim.x + threadIdx.x;
    if (i >= n4) return;
    float4 v = ((const float4*)x)[i];
    ushort4 o;
    o.x = f2bf_rne(v.x);
    o.y = f2bf_rne(v.y);
    o.z = f2bf_rne(v.z);
    o.w = f2bf_rne(v.w);
    ((ushort4*)xb)[i] = o;
}

// ---------------------------------------------------------------------------
// Kernel 2: build w_bf16[O,I] from logits[O,I,2], scales[O,I/128], mask[O,I]
// ---------------------------------------------------------------------------
__global__ __launch_bounds__(256) void build_w_kernel(const float* __restrict__ logits,
                                                      const float* __restrict__ scales,
                                                      const int*   __restrict__ mask,
                                                      unsigned short* __restrict__ wb,
                                                      int O, int I, int groups) {
    int t = blockIdx.x * blockDim.x + threadIdx.x;
    int n4 = (O * I) >> 2;
    if (t >= n4) return;
    int idx = t << 2;
    int o = idx / I;
    int i = idx - o * I;

    float s = bf2f(f2bf_rne(scales[(size_t)o * groups + (i >> 7)]));

    const float4* lg = (const float4*)(logits + ((size_t)idx << 1));
    float4 p01 = lg[0];
    float4 p23 = lg[1];
    int4 mk = *(const int4*)(mask + idx);

    float l0[4] = {p01.x, p01.z, p23.x, p23.z};
    float l1[4] = {p01.y, p01.w, p23.y, p23.w};
    int   mm[4] = {mk.x, mk.y, mk.z, mk.w};

    unsigned short r[4];
#pragma unroll
    for (int e = 0; e < 4; ++e) {
        float a = l0[e], b = l1[e];
        float m  = fmaxf(fmaxf(a, b), 0.0f);
        float e0 = __expf(a - m);
        float e1 = __expf(b - m);
        float ez = __expf(-m);
        float w  = (e1 - e0) / (e0 + ez + e1);
        w *= s;
        w *= (float)mm[e];
        r[e] = f2bf_rne(w);
    }
    ushort4 out;
    out.x = r[0]; out.y = r[1]; out.z = r[2]; out.w = r[3];
    ((ushort4*)wb)[t] = out;
}

// ---------------------------------------------------------------------------
// Kernel 3: out[T,O] = x_bf16[T,I] @ w_bf16[O,I]^T
//
// Block: 256 threads = 8 waves as 2(M) x 4(N); block tile 128x256.
// K-stage 64, double-buffered LDS (padded 144B row stride), async copies.
//
// Fragment layouts (contiguous K, same for A and B), served by 16B ds loads:
//   A 16x32 bf16: lane<16 -> row M=lane,   K 0..15;          lane>=16 -> K 16..31
//   B 32x16 bf16: lane<16 -> col N=lane,   K 0..15;          lane>=16 -> K 16..31
// ---------------------------------------------------------------------------
#define ROWB     144                      // padded LDS row stride (bytes)
#define A_BYTES  (128 * ROWB)             // 18432
#define B_BYTES  (256 * ROWB)             // 36864
#define STAGE    (A_BYTES + B_BYTES)      // 55296
#define SMEM_TOT (2 * STAGE)              // 110592

extern __shared__ uint4 dynsmem[];

__global__ __launch_bounds__(256) void gemm_bf16_lds_kernel(const unsigned short* __restrict__ xb,
                                                            const unsigned short* __restrict__ wb,
                                                            float* __restrict__ out,
                                                            int T, int O, int I) {
    const int tid    = threadIdx.x;
    const int lane   = tid & 31;
    const int wid    = tid >> 5;          // 0..7
    const int waveM  = wid >> 2;          // 0..1
    const int waveN  = wid & 3;           // 0..3

    const int gnTiles = O >> 8;           // O/256
    const int bm = blockIdx.x / gnTiles;
    const int bn = blockIdx.x - bm * gnTiles;
    const int m0 = bm << 7;               // block row base
    const int n0 = bn << 8;               // block col base

    const int laneHi = lane >> 4;
    const int lane15 = lane & 15;
    const int aOff = laneHi << 4;         // +16 halves (A hi-lane K offset)
    const int bOff = laneHi << 4;         // +16 halves (B hi-lane K offset)

    char* smemc = (char*)dynsmem;
    const unsigned smemBase = (unsigned)(uintptr_t)smemc;  // LDS byte address = low 32 bits

    // per-thread async-copy assignment
    const int aRow   = tid >> 1;                 // 0..127
    const int aInRow = (tid & 1) << 6;           // 0 or 64 bytes
    const int bCol   = tid;                      // 0..255

    // stage loader: copy A[128 x 64] and B[256 x 64] bf16 tiles for k0
    auto stage_load = [&](int k0, int buf) {
        unsigned ldsA = smemBase + (unsigned)(buf * STAGE) + (unsigned)(aRow * ROWB) + (unsigned)aInRow;
        const char* gA = (const char*)(xb + (size_t)(m0 + aRow) * I + k0) + aInRow;
        async_copy64(ldsA, gA);                  // 4 async b128
        unsigned ldsB = smemBase + (unsigned)(buf * STAGE + A_BYTES) + (unsigned)(bCol * ROWB);
        const char* gB = (const char*)(wb + (size_t)(n0 + bCol) * I + k0);
        async_copy128(ldsB, gB);                 // 8 async b128
    };

    v8f acc[4][4] = {};

    const int nStages = I >> 6;                  // I/64
    stage_load(0, 0);

    for (int s = 0; s < nStages; ++s) {
        const int cur = s & 1;
        if (s + 1 < nStages) {
            stage_load((s + 1) << 6, cur ^ 1);
            WAIT_ASYNC(12);                      // previous stage's 12 ops done
        } else {
            WAIT_ASYNC(0);
        }
        __syncthreads();                         // stage s visible to all waves

        const unsigned short* As = (const unsigned short*)(smemc + cur * STAGE);
        const unsigned short* Bs = (const unsigned short*)(smemc + cur * STAGE + A_BYTES);

#pragma unroll
        for (int kk = 0; kk < 64; kk += 32) {
            BFrag b[4];
#pragma unroll
            for (int ni = 0; ni < 4; ++ni) {
                const unsigned short* q =
                    Bs + (size_t)((waveN << 6) + (ni << 4) + lane15) * (ROWB / 2) + kk + bOff;
                b[ni].u[0] = *(const uint4*)(q);
                b[ni].u[1] = *(const uint4*)(q + 8);
            }
#pragma unroll
            for (int mi = 0; mi < 4; ++mi) {
                BFrag a;
                const unsigned short* p =
                    As + (size_t)((waveM << 6) + (mi << 4) + lane15) * (ROWB / 2) + kk + aOff;
                a.u[0] = *(const uint4*)(p);
                a.u[1] = *(const uint4*)(p + 8);
#pragma unroll
                for (int ni = 0; ni < 4; ++ni) {
                    acc[mi][ni] = __builtin_amdgcn_wmma_f32_16x16x32_bf16(
                        false, a.v, false, b[ni].v,
                        (short)0, acc[mi][ni], false, false);
                }
            }
        }
        __syncthreads();                         // done reading buf before it is refilled
    }

    // C/D layout: VGPR r -> row (r + 8*laneHi), lanes stripe N.
    const int mw = m0 + (waveM << 6);
    const int nw = n0 + (waveN << 6);
#pragma unroll
    for (int mi = 0; mi < 4; ++mi) {
#pragma unroll
        for (int ni = 0; ni < 4; ++ni) {
#pragma unroll
            for (int r = 0; r < 8; ++r) {
                int row = mw + (mi << 4) + r + (laneHi << 3);
                int col = nw + (ni << 4) + lane15;
                out[(size_t)row * O + col] = acc[mi][ni][r];
            }
        }
    }
}

// ---------------------------------------------------------------------------
// Host launcher
// ---------------------------------------------------------------------------
extern "C" void kernel_launch(void* const* d_in, const int* in_sizes, int n_in,
                              void* d_out, int out_size, void* d_ws, size_t ws_size,
                              hipStream_t stream) {
    const float* x      = (const float*)d_in[0];   // [T, I]
    const float* logits = (const float*)d_in[1];   // [O, I, 2]
    const float* scales = (const float*)d_in[2];   // [O, I/128]
    const int*   mask   = (const int*)d_in[3];     // [O, I]
    float*       out    = (float*)d_out;           // [T, O]

    const int I = 4096;
    const int O = in_sizes[3] / I;                 // 4096
    const int T = in_sizes[0] / I;                 // 8192
    const int groups = I / 128;

    unsigned short* xb = (unsigned short*)d_ws;            // T*I bf16 (64 MB)
    unsigned short* wb = xb + (size_t)T * I;               // O*I bf16 (32 MB)

    {   // x -> bf16
        int n4 = (T * I) >> 2;
        cvt_x_kernel<<<(n4 + 255) / 256, 256, 0, stream>>>(x, xb, n4);
    }
    {   // ternary-softmax weights in bf16
        int n4 = (O * I) >> 2;
        build_w_kernel<<<(n4 + 255) / 256, 256, 0, stream>>>(logits, scales, mask, wb, O, I, groups);
    }
    {   // WMMA GEMM: 128x256 block tile, 8 waves, 108KB dynamic LDS
        (void)hipFuncSetAttribute((const void*)gemm_bf16_lds_kernel,
                                  hipFuncAttributeMaxDynamicSharedMemorySize, SMEM_TOT);
        int blocks = (T >> 7) * (O >> 8);          // 64 * 16 = 1024
        gemm_bf16_lds_kernel<<<blocks, 256, SMEM_TOT, stream>>>(xb, wb, out, T, O, I);
    }
}
